// CausalSelfAttention_23244363005983
// MI455X (gfx1250) — matrix operand order repaired and checked by reading, not compiled, not run on hardware
//
#include <hip/hip_runtime.h>
#include <hip/hip_bf16.h>

// CausalSelfAttention with ALiBi — CDNA5 (gfx1250) WMMA implementation.
// B=2, T=2048, C=1024, NH=16, HD=64.
// All WMMA operands are staged in LDS with the contraction dim innermost so
// every fragment load is 2x ds_load_b128 (no scalar u16 gathers).

#define Bsz  2
#define Tsz  2048
#define Csz  1024
#define NHsz 16
#define HDsz 64

typedef __attribute__((ext_vector_type(16))) _Float16 v16h;
typedef __attribute__((ext_vector_type(8)))  _Float16 v8h;
typedef __attribute__((ext_vector_type(8)))  float    v8f;

__device__ __forceinline__ int lane_id() { return (int)(threadIdx.x & 31); }

__device__ __forceinline__ v8f vzero8() {
    v8f z = {0.f, 0.f, 0.f, 0.f, 0.f, 0.f, 0.f, 0.f};
    return z;
}

// ---------------------------------------------------------------------------
// WMMA fragment loaders per CDNA5 ISA 7.12.2 (wave32), vectorized.
// A: 16(M) x 32(K), source row-major [M][K] (contraction contiguous per row).
// Lane l (m=l&15, half=l>>4) holds halves [half*8, +8) and [16+half*8, +8).
// Row byte stride (2*ld) must be a multiple of 16 for alignment.
// ---------------------------------------------------------------------------
__device__ __forceinline__ v16h frag_a(const _Float16* p, int ld) {
    const int l = lane_id();
    const int m = l & 15, half = l >> 4;
    const _Float16* base = p + (size_t)m * ld + half * 8;
    const v8h lo = *(const v8h*)(base);
    const v8h hi = *(const v8h*)(base + 16);
    return __builtin_shufflevector(lo, hi, 0, 1, 2, 3, 4, 5, 6, 7,
                                   8, 9, 10, 11, 12, 13, 14, 15);
}

// B: 32(K) x 16(N), source stored [N][K] (contraction contiguous per row).
// Lane l (n=l&15, half=l>>4) holds halves [half*8, +8) and [16+half*8, +8) of row n.
__device__ __forceinline__ v16h frag_b_ct(const _Float16* p, int ld) {
    const int l = lane_id();
    const int n = l & 15, half = l >> 4;
    const _Float16* base = p + (size_t)n * ld + half * 8;
    const v8h lo = *(const v8h*)(base);
    const v8h hi = *(const v8h*)(base + 16);
    return __builtin_shufflevector(lo, hi, 0, 1, 2, 3, 4, 5, 6, 7,
                                   8, 9, 10, 11, 12, 13, 14, 15);
}

// ---------------------------------------------------------------------------
// Tiled GEMM with WMMA, double-buffered LDS.  out[M,N] = A*W + bias.
// Block = 256 threads (8 waves), tile 128x128, BK = 32.
// Wave (wm = w&3, wn = w>>2) owns a 32(M) x 64(N) output slab: 8 WMMA / K-step.
// W is staged TRANSPOSED in LDS ([n][k]) so B-fragments are b128 loads.
// MODE 0: A=f32 x, write q  -> f16 [B*NH, T, HD]
// MODE 1: A=f32 x, write kv -> f16 k/v [B*NH, T, HD]
// MODE 2: A=f16 y, write f32 row-major d_out
// ---------------------------------------------------------------------------
template <int MODE>
__global__ void __launch_bounds__(256)
gemm_wmma(const float* __restrict__ Af32, const _Float16* __restrict__ Af16,
          const float* __restrict__ W, const float* __restrict__ bias,
          _Float16* __restrict__ out_h, _Float16* __restrict__ out_h2,
          float* __restrict__ out_f, int M, int K, int N) {
    __shared__ _Float16 As[2][128][40];    // [m][k], 128 x 32 (+8 pad)
    __shared__ _Float16 BsT[2][128][40];   // [n][k], 128 x 32 (+8 pad)

    const int tid = threadIdx.x;
    const int w   = tid >> 5;
    const int wm  = w & 3;          // M quadrant (32 rows)
    const int wn  = w >> 2;         // N half (64 cols)
    const int m0  = blockIdx.x * 128;
    const int n0  = blockIdx.y * 128;

    const int ar = tid >> 1, ac = (tid & 1) * 16;       // A: 16 elems/thread
    const int bn = tid & 127, bk = (tid >> 7) * 16;     // W: 16 elems/thread (col)

    float a_st[16], w_st[16];

    auto fetch = [&](int k0) {
        if (MODE == 2) {
            const _Float16* src = Af16 + (size_t)(m0 + ar) * K + k0 + ac;
#pragma unroll
            for (int i = 0; i < 16; ++i) a_st[i] = (float)src[i];
        } else {
            const float* src = Af32 + (size_t)(m0 + ar) * K + k0 + ac;
#pragma unroll
            for (int i = 0; i < 16; ++i) a_st[i] = src[i];
        }
        // Column of W: coalesced across the wave (consecutive n per lane).
#pragma unroll
        for (int i = 0; i < 16; ++i)
            w_st[i] = W[(size_t)(k0 + bk + i) * N + n0 + bn];
    };
    auto commit = [&](int buf) {
#pragma unroll
        for (int i = 0; i < 16; ++i) As[buf][ar][ac + i] = (_Float16)a_st[i];
#pragma unroll
        for (int i = 0; i < 16; ++i) BsT[buf][bn][bk + i] = (_Float16)w_st[i];
    };

    v8f acc[2][4];
#pragma unroll
    for (int s = 0; s < 2; ++s)
#pragma unroll
        for (int nt = 0; nt < 4; ++nt) acc[s][nt] = vzero8();

    fetch(0);
    commit(0);
    __syncthreads();

    const int nK = K / 32;
    for (int it = 0; it < nK; ++it) {
        const int cur = it & 1;
        if (it + 1 < nK) fetch((it + 1) * 32);   // overlap global with WMMA

        const v16h a0 = frag_a(&As[cur][wm * 32 + 0][0], 40);
        const v16h a1 = frag_a(&As[cur][wm * 32 + 16][0], 40);
#pragma unroll
        for (int nt = 0; nt < 4; ++nt) {
            const v16h bf = frag_b_ct(&BsT[cur][wn * 64 + nt * 16][0], 40);
            acc[0][nt] = __builtin_amdgcn_wmma_f32_16x16x32_f16(
                false, a0, false, bf, (short)0, acc[0][nt], false, false);
            acc[1][nt] = __builtin_amdgcn_wmma_f32_16x16x32_f16(
                false, a1, false, bf, (short)0, acc[1][nt], false, false);
        }

        if (it + 1 < nK) commit((it + 1) & 1);
        __syncthreads();
    }

    // Epilogue: bias + scatter per destination layout.
    const int l = lane_id();
    const int nl = l & 15, half = l >> 4;
#pragma unroll
    for (int sub = 0; sub < 2; ++sub) {
#pragma unroll
        for (int nt = 0; nt < 4; ++nt) {
            const int n  = n0 + wn * 64 + nt * 16 + nl;
            const float bv = bias[n];
#pragma unroll
            for (int i = 0; i < 8; ++i) {
                const int m    = m0 + wm * 32 + sub * 16 + half * 8 + i;
                const float val = acc[sub][nt][i] + bv;
                if (MODE == 2) {
                    out_f[(size_t)m * N + n] = val;
                } else if (MODE == 0) {
                    const int bb = m / Tsz, t = m % Tsz;
                    const int h = n >> 6, d = n & 63;
                    out_h[(((size_t)(bb * NHsz + h)) * Tsz + t) * HDsz + d] = (_Float16)val;
                } else {
                    const int bb = m / Tsz, t = m % Tsz;
                    const int kvsel = n >> 10;            // 0 -> k, 1 -> v
                    const int h = (n >> 6) & 15, d = n & 63;
                    _Float16* dst = kvsel ? out_h2 : out_h;
                    dst[(((size_t)(bb * NHsz + h)) * Tsz + t) * HDsz + d] = (_Float16)val;
                }
            }
        }
    }
}

// ---------------------------------------------------------------------------
// Flash attention with ALiBi + causal mask, double-buffered K/V stream.
// Grid: (T/64, B*NH). Block = 128 threads (4 waves); each wave owns 16 q rows.
// 1/sqrt(HD) is folded into the staged Q tile. V is staged TRANSPOSED
// ([hd][key]) so the PV B-fragments are b128 loads.
// ---------------------------------------------------------------------------
__global__ void __launch_bounds__(128)
attn_wmma(const _Float16* __restrict__ qf, const _Float16* __restrict__ kf,
          const _Float16* __restrict__ vf, _Float16* __restrict__ yf) {
    __shared__ _Float16 Qs[64][72];        // [qrow][hd]
    __shared__ _Float16 Ks[2][64][72];     // [key][hd]   (S:  contraction = hd)
    __shared__ _Float16 VsT[2][64][72];    // [hd][key]   (PV: contraction = key)
    __shared__ _Float16 Ps[4][16][72];     // [qrow][key] per wave

    const int tid = threadIdx.x;
    const int w   = tid >> 5;
    const int l   = lane_id();
    const int nl  = l & 15, half = l >> 4;

    const int bh = blockIdx.y;
    const int b  = bh / NHsz;
    const int h  = bh % NHsz;
    const int q0 = blockIdx.x * 64;

    const float slope = (float)(h + 1) / 16.0f;    // ALiBi slope

    const _Float16* Q  = qf + (size_t)bh * Tsz * HDsz;
    const _Float16* Km = kf + (size_t)bh * Tsz * HDsz;
    const _Float16* Vm = vf + (size_t)bh * Tsz * HDsz;

    const int r  = tid >> 1, c0 = (tid & 1) * 32;  // K staging (row-major)
    const int vd = tid & 63, vk = (tid >> 6) * 32; // V staging (transposed)

    float4   kreg[4];
    _Float16 varr[32];
    auto fetch_kv = [&](int j) {
        const float4* ks = (const float4*)(Km + (size_t)(j + r) * HDsz + c0);
#pragma unroll
        for (int i = 0; i < 4; ++i) kreg[i] = ks[i];
        // Column of V: coalesced across the wave (consecutive hd per lane).
#pragma unroll
        for (int i = 0; i < 32; ++i)
            varr[i] = Vm[(size_t)(j + vk + i) * HDsz + vd];
    };
    auto commit_kv = [&](int buf) {
        float4* kd = (float4*)&Ks[buf][r][c0];
#pragma unroll
        for (int i = 0; i < 4; ++i) kd[i] = kreg[i];
#pragma unroll
        for (int i = 0; i < 32; ++i) VsT[buf][vd][vk + i] = varr[i];
    };

    // Stage Q tile once, folding in 1/sqrt(HD).
    {
        const _Float16* src = Q + (size_t)(q0 + r) * HDsz + c0;
#pragma unroll
        for (int i = 0; i < 32; ++i)
            Qs[r][c0 + i] = (_Float16)((float)src[i] * 0.125f);
    }
    fetch_kv(0);
    commit_kv(0);
    __syncthreads();

    const v16h qa0 = frag_a(&Qs[w * 16][0], 72);
    const v16h qa1 = frag_a(&Qs[w * 16][32], 72);

    float m_i[8], l_i[8];
#pragma unroll
    for (int i = 0; i < 8; ++i) { m_i[i] = -3.0e38f; l_i[i] = 0.0f; }
    v8f o_acc[4];
#pragma unroll
    for (int nt = 0; nt < 4; ++nt) o_acc[nt] = vzero8();

    const int qrow_base = q0 + w * 16 + half * 8;
    const int nblocks   = q0 / 64 + 1;

    for (int it = 0; it < nblocks; ++it) {
        const int cur = it & 1;
        const int j0  = it * 64;

        if (it + 1 < nblocks) fetch_kv((it + 1) * 64);   // overlap next K/V
        if (it + 2 < nblocks) {                          // prefetch 2 ahead
            __builtin_prefetch(Km + (size_t)((it + 2) * 64 + r) * HDsz + c0, 0, 1);
            __builtin_prefetch(Vm + (size_t)((it + 2) * 64 + r) * HDsz + c0, 0, 1);
        }

        // S = (Q/sqrt(hd)) * K^T : contraction over HD (2 chunks of 32).
        v8f s[4];
#pragma unroll
        for (int nt = 0; nt < 4; ++nt) {
            s[nt] = vzero8();
            const v16h kb0 = frag_b_ct(&Ks[cur][nt * 16][0], 72);
            s[nt] = __builtin_amdgcn_wmma_f32_16x16x32_f16(
                false, qa0, false, kb0, (short)0, s[nt], false, false);
            const v16h kb1 = frag_b_ct(&Ks[cur][nt * 16][32], 72);
            s[nt] = __builtin_amdgcn_wmma_f32_16x16x32_f16(
                false, qa1, false, kb1, (short)0, s[nt], false, false);
        }

        // ALiBi + causal mask (scale already folded into Q).
#pragma unroll
        for (int nt = 0; nt < 4; ++nt) {
#pragma unroll
            for (int i = 0; i < 8; ++i) {
                const int qi = qrow_base + i;
                const int kj = j0 + nt * 16 + nl;
                const float sv = s[nt][i] + slope * (float)(kj - qi);
                s[nt][i] = (kj > qi) ? -3.0e38f : sv;
            }
        }

        // Online softmax update (row reductions across the 16-lane half).
        float fac[8];
#pragma unroll
        for (int i = 0; i < 8; ++i) {
            float mx = fmaxf(fmaxf(s[0][i], s[1][i]), fmaxf(s[2][i], s[3][i]));
#pragma unroll
            for (int off = 1; off < 16; off <<= 1)
                mx = fmaxf(mx, __shfl_xor(mx, off, 32));
            const float mnew = fmaxf(m_i[i], mx);
            fac[i] = __expf(m_i[i] - mnew);
            m_i[i] = mnew;
            l_i[i] *= fac[i];
        }

#pragma unroll
        for (int i = 0; i < 8; ++i) {
            float rs = 0.0f;
#pragma unroll
            for (int nt = 0; nt < 4; ++nt) {
                const float p = __expf(s[nt][i] - m_i[i]);
                rs += p;
                Ps[w][half * 8 + i][nt * 16 + nl] = (_Float16)p;
            }
#pragma unroll
            for (int off = 1; off < 16; off <<= 1)
                rs += __shfl_xor(rs, off, 32);
            l_i[i] += rs;
#pragma unroll
            for (int nt = 0; nt < 4; ++nt) o_acc[nt][i] *= fac[i];
        }
        __syncthreads();   // Ps visible; K/V buffer cur no longer written

        if (it + 1 < nblocks) commit_kv((it + 1) & 1);   // fill other buffer

        // O += P * V : contraction over 64 keys (2 chunks of 32).
        const v16h pa0 = frag_a(&Ps[w][0][0], 72);
        const v16h pa1 = frag_a(&Ps[w][0][32], 72);
#pragma unroll
        for (int nt = 0; nt < 4; ++nt) {
            const v16h vb0 = frag_b_ct(&VsT[cur][nt * 16][0], 72);
            o_acc[nt] = __builtin_amdgcn_wmma_f32_16x16x32_f16(
                false, pa0, false, vb0, (short)0, o_acc[nt], false, false);
            const v16h vb1 = frag_b_ct(&VsT[cur][nt * 16][32], 72);
            o_acc[nt] = __builtin_amdgcn_wmma_f32_16x16x32_f16(
                false, pa1, false, vb1, (short)0, o_acc[nt], false, false);
        }
        __syncthreads();   // commit visible for next S; PV reads done
    }

    // Normalize and write y as f16 [B*T, C] for the output projection.
#pragma unroll
    for (int nt = 0; nt < 4; ++nt) {
#pragma unroll
        for (int i = 0; i < 8; ++i) {
            const int t   = q0 + w * 16 + half * 8 + i;
            const size_t row = (size_t)b * Tsz + t;
            const int col = h * HDsz + nt * 16 + nl;
            yf[row * Csz + col] = (_Float16)(o_acc[nt][i] / l_i[i]);
        }
    }
}

// ---------------------------------------------------------------------------
extern "C" void kernel_launch(void* const* d_in, const int* in_sizes, int n_in,
                              void* d_out, int out_size, void* d_ws, size_t ws_size,
                              hipStream_t stream) {
    (void)in_sizes; (void)n_in; (void)out_size; (void)ws_size;

    const float* x    = (const float*)d_in[0];
    const float* q_w  = (const float*)d_in[1];
    const float* q_b  = (const float*)d_in[2];
    const float* kv_w = (const float*)d_in[3];
    const float* kv_b = (const float*)d_in[4];
    const float* o_w  = (const float*)d_in[5];
    const float* o_b  = (const float*)d_in[6];
    float* out = (float*)d_out;

    const size_t per_tensor = (size_t)Bsz * NHsz * Tsz * HDsz;  // 4M f16 = 8 MB
    _Float16* qf = (_Float16*)d_ws;
    _Float16* kf = qf + per_tensor;
    _Float16* vf = kf + per_tensor;
    _Float16* yf = vf + per_tensor;   // [B*T, C]

    const int M = Bsz * Tsz;          // 4096
    dim3 blk256(256), blk128(128);

    // q projection: N = 1024
    gemm_wmma<0><<<dim3(M / 128, Csz / 128), blk256, 0, stream>>>(
        x, nullptr, q_w, q_b, qf, nullptr, nullptr, M, Csz, Csz);
    // kv projection: N = 2048
    gemm_wmma<1><<<dim3(M / 128, (2 * Csz) / 128), blk256, 0, stream>>>(
        x, nullptr, kv_w, kv_b, kf, vf, nullptr, M, Csz, 2 * Csz);
    // flash attention with ALiBi
    attn_wmma<<<dim3(Tsz / 64, Bsz * NHsz), blk128, 0, stream>>>(qf, kf, vf, yf);
    // output projection: f32 out
    gemm_wmma<2><<<dim3(M / 128, Csz / 128), blk256, 0, stream>>>(
        nullptr, yf, o_w, o_b, nullptr, nullptr, out, M, Csz, Csz);
}
